// WindowedCrossAttn_14645838479940
// MI455X (gfx1250) — matrix operand order repaired and checked by reading, not compiled, not run on hardware
//
#include <hip/hip_runtime.h>

// Problem constants
#define Bc   2
#define Sc   2048
#define Dc   1024
#define Hc   16
#define DHc  64
#define WINc 256
#define SPc  2112           // padded key stride for V^T buffer (S + 64)
#define Mc   (Bc * Sc)      // 4096 tokens

typedef __attribute__((ext_vector_type(16))) __bf16 v16bf;
typedef __attribute__((ext_vector_type(8)))  float  v8f;

__device__ __forceinline__ v8f wmma_bf16(v16bf a, v16bf b, v8f c) {
  // v_wmma_f32_16x16x32_bf16  (8-arg form: neg_a, A, neg_b, B, c_mod, C, reuse_a, reuse_b)
  return __builtin_amdgcn_wmma_f32_16x16x32_bf16(false, a, false, b, (short)0, c, false, false);
}

__device__ __forceinline__ v8f zero_v8f() {
  v8f z;
#pragma unroll
  for (int e = 0; e < 8; ++e) z[e] = 0.0f;
  return z;
}

// ---- A operand (16x32, bf16): lane = row (lane&15); lanes 0-15 hold K=0..15, lanes 16-31 hold K=16..31 (contiguous).
__device__ __forceinline__ v16bf load_a_bf16(const __bf16* base, int ld, int lane) {
  const int row = lane & 15;
  const int kb  = (lane >> 4) << 4;
  const __bf16* p = base + row * ld + kb;
  v16bf f;
#pragma unroll
  for (int e = 0; e < 8; ++e) f[e] = p[e];
#pragma unroll
  for (int e = 0; e < 8; ++e) f[8 + e] = p[8 + e];
  return f;
}

// ---- B operand (32x16, bf16), sourced from B^T stored row-major (N rows, K cols):
//   lane = n (lane&15); lanes 0-15 hold K=0..15, lanes 16-31 hold K=16..31 (contiguous).
__device__ __forceinline__ v16bf load_bt_bf16(const __bf16* bt, int ld, int lane) {
  const int n  = lane & 15;
  const int kk = (lane >> 4) << 4;
  const __bf16* p = bt + n * ld + kk;
  v16bf f;
#pragma unroll
  for (int e = 0; e < 16; ++e) f[e] = p[e];
  return f;
}

// ---------------- fp32 -> bf16 convert (weights & activations) ----------------
__global__ void cvt_f32_bf16_kernel(const float* __restrict__ src,
                                    __bf16* __restrict__ dst, int n) {
  int i = blockIdx.x * blockDim.x + threadIdx.x;
  if (i < n) dst[i] = (__bf16)src[i];
}

// ---------------- unified GEMM: out = X @ W^T + b ----------------
// One wave computes a 32(M) x 64(N) tile: 2 A frags x 4 B frags -> 8 WMMAs / k-step.
// MODE 0: bf16 head-major [B,H,S,DH]; MODE 2: bf16 V^T [B,H,DH,SPc]; MODE 3: fp32 row-major.
template <int MODE>
__global__ void __launch_bounds__(128)
gemm_kernel(const __bf16* __restrict__ X, const __bf16* __restrict__ Wbf,
            const float* __restrict__ bias, __bf16* __restrict__ outb,
            float* __restrict__ outf) {
  const int lane = threadIdx.x & 31;
  const int wave = threadIdx.x >> 5;
  const int row0 = (blockIdx.x * 4 + wave) * 32;   // token tile (2 x 16 rows)
  const int n0   = blockIdx.y * 64;                // output-feature strip

  v8f acc0[4], acc1[4];
#pragma unroll
  for (int t = 0; t < 4; ++t) { acc0[t] = zero_v8f(); acc1[t] = zero_v8f(); }

  for (int k = 0; k < Dc; k += 32) {
    const v16bf a0 = load_a_bf16(X + row0 * Dc + k, Dc, lane);
    const v16bf a1 = load_a_bf16(X + (row0 + 16) * Dc + k, Dc, lane);
#pragma unroll
    for (int t = 0; t < 4; ++t) {
      const v16bf b = load_bt_bf16(Wbf + (n0 + t * 16) * Dc + k, Dc, lane);
      acc0[t] = wmma_bf16(a0, b, acc0[t]);
      acc1[t] = wmma_bf16(a1, b, acc1[t]);
    }
  }

  const int nl = lane & 15;
  const int mb = (lane >> 4) << 3;
#pragma unroll
  for (int t = 0; t < 4; ++t) {
    const int n  = n0 + t * 16 + nl;
    const float bv = bias[n];
    const int h = n >> 6, dh = n & 63;
#pragma unroll
    for (int half = 0; half < 2; ++half) {
#pragma unroll
      for (int r = 0; r < 8; ++r) {
        const int m  = row0 + half * 16 + mb + r;
        const float val = (half ? acc1[t][r] : acc0[t][r]) + bv;
        if (MODE == 3) {
          outf[m * Dc + n] = val;
        } else {
          const int bb = m >> 11, s = m & (Sc - 1);
          const int addr = (MODE == 2) ? ((bb * Hc + h) * DHc + dh) * SPc + s
                                       : ((bb * Hc + h) * Sc + s) * DHc + dh;
          outb[addr] = (__bf16)val;
        }
      }
    }
  }
}

// ---------------- windowed flash attention ----------------
// grid (B*H, S/16); one wave per 16-query tile.
__global__ void __launch_bounds__(32)
attn_kernel(const __bf16* __restrict__ qm, const __bf16* __restrict__ km,
            const __bf16* __restrict__ vt, __bf16* __restrict__ ctx) {
  __shared__ __bf16 pl[16 * 32];   // P tile staging (C-layout -> A-layout)

  const int lane = threadIdx.x & 31;
  const int bh   = blockIdx.x;               // b*H + h
  const int b    = bh >> 4, h = bh & 15;
  const int q0   = blockIdx.y * 16;
  const int nl   = lane & 15;
  const int mb   = (lane >> 4) << 3;
  const float scale = 0.125f;                // 1/sqrt(64)

  const long qoff  = ((long)bh * Sc + q0) * DHc;
  const long koff  = (long)bh * Sc * DHc;
  const long vtoff = (long)bh * DHc * SPc;

  const v16bf qa0 = load_a_bf16(qm + qoff +  0, DHc, lane);
  const v16bf qa1 = load_a_bf16(qm + qoff + 32, DHc, lane);

  v8f o[4];
#pragma unroll
  for (int t = 0; t < 4; ++t) o[t] = zero_v8f();
  float mrow[8], lrow[8];
#pragma unroll
  for (int r = 0; r < 8; ++r) { mrow[r] = -3.0e38f; lrow[r] = 0.0f; }

  int lo = q0 - (WINc - 1); if (lo < 0) lo = 0;
  const int cb0 = lo & ~31;

  for (int cb = cb0; cb <= q0 + 15; cb += 32) {
    // ---- scores: two 16x16 tiles over 32 keys, K-dim = DH in two chunks of 32
    v8f s0 = zero_v8f(), s1 = zero_v8f();
    {
      int key0 = cb + nl;       if (key0 > Sc - 1) key0 = Sc - 1;
      int key1 = cb + 16 + nl;  if (key1 > Sc - 1) key1 = Sc - 1;
      const int dsel = (lane >> 4) << 4;
      const __bf16* kp0 = km + koff + (long)key0 * DHc + dsel;
      const __bf16* kp1 = km + koff + (long)key1 * DHc + dsel;
      v16bf k00, k01, k10, k11;
#pragma unroll
      for (int e = 0; e < 16; ++e) {
        k00[e] = kp0[e]; k01[e] = kp0[32 + e];
        k10[e] = kp1[e]; k11[e] = kp1[32 + e];
      }
      s0 = wmma_bf16(qa0, k00, s0); s0 = wmma_bf16(qa1, k01, s0);
      s1 = wmma_bf16(qa0, k10, s1); s1 = wmma_bf16(qa1, k11, s1);
    }

    // ---- online softmax (row reductions across 16-lane groups)
    float al[8];
#pragma unroll
    for (int r = 0; r < 8; ++r) {
      const int qi  = q0 + mb + r;
      const int k0i = cb + nl, k1i = k0i + 16;
      const bool keep0 = (k0i <= qi) && (k0i >= qi - (WINc - 1));
      const bool keep1 = (k1i <= qi) && (k1i >= qi - (WINc - 1));
      float v0 = keep0 ? s0[r] * scale : -3.0e38f;
      float v1 = keep1 ? s1[r] * scale : -3.0e38f;
      float mx = fmaxf(v0, v1);
      mx = fmaxf(mx, __shfl_xor(mx, 1, 32));
      mx = fmaxf(mx, __shfl_xor(mx, 2, 32));
      mx = fmaxf(mx, __shfl_xor(mx, 4, 32));
      mx = fmaxf(mx, __shfl_xor(mx, 8, 32));
      mx = fmaxf(mx, mrow[r]);
      const float a  = (mrow[r] > -1.0e38f) ? __expf(mrow[r] - mx) : 0.0f;
      const float e0 = keep0 ? __expf(v0 - mx) : 0.0f;
      const float e1 = keep1 ? __expf(v1 - mx) : 0.0f;
      float sum = e0 + e1;
      sum += __shfl_xor(sum, 1, 32);
      sum += __shfl_xor(sum, 2, 32);
      sum += __shfl_xor(sum, 4, 32);
      sum += __shfl_xor(sum, 8, 32);
      lrow[r] = lrow[r] * a + sum;
      mrow[r] = mx;
      al[r]   = a;
      pl[(mb + r) * 32 + nl]      = (__bf16)e0;
      pl[(mb + r) * 32 + 16 + nl] = (__bf16)e1;
    }

    // ---- P (via LDS transpose) x V^T strips
    const v16bf pa = load_a_bf16(pl, 32, lane);
#pragma unroll
    for (int t = 0; t < 4; ++t) {
#pragma unroll
      for (int r = 0; r < 8; ++r) o[t][r] *= al[r];
    }
    const int ksel = (lane >> 4) << 4;
#pragma unroll
    for (int t = 0; t < 4; ++t) {
      const __bf16* vp = vt + vtoff + (long)(t * 16 + nl) * SPc + cb + ksel;
      v16bf vb;
#pragma unroll
      for (int e = 0; e < 16; ++e) vb[e] = vp[e];
      o[t] = wmma_bf16(pa, vb, o[t]);
    }
  }

  // ---- normalize + write ctx row-major [B*S, D] (bf16)
#pragma unroll
  for (int r = 0; r < 8; ++r) lrow[r] = (lrow[r] > 0.0f) ? 1.0f / lrow[r] : 0.0f;
  const long crow0 = (long)b * Sc + q0;
#pragma unroll
  for (int t = 0; t < 4; ++t) {
#pragma unroll
    for (int r = 0; r < 8; ++r) {
      const long addr = (crow0 + mb + r) * Dc + h * DHc + t * 16 + nl;
      ctx[addr] = (__bf16)(o[t][r] * lrow[r]);
    }
  }
}

extern "C" void kernel_launch(void* const* d_in, const int* in_sizes, int n_in,
                              void* d_out, int out_size, void* d_ws, size_t ws_size,
                              hipStream_t stream) {
  (void)in_sizes; (void)n_in; (void)out_size; (void)ws_size;

  const float* Xq = (const float*)d_in[0];
  const float* Xk = (const float*)d_in[1];
  const float* Xv = (const float*)d_in[2];
  const float* Wq = (const float*)d_in[3];
  const float* bq = (const float*)d_in[4];
  const float* Wk = (const float*)d_in[5];
  const float* bk = (const float*)d_in[6];
  const float* Wv = (const float*)d_in[7];
  const float* bv = (const float*)d_in[8];
  const float* Wo = (const float*)d_in[9];
  const float* bo = (const float*)d_in[10];

  char* ws = (char*)d_ws;
  const size_t MB = (size_t)1 << 20;
  __bf16* wqb = (__bf16*)(ws + 0 * MB);
  __bf16* wkb = (__bf16*)(ws + 2 * MB);
  __bf16* wvb = (__bf16*)(ws + 4 * MB);
  __bf16* wob = (__bf16*)(ws + 6 * MB);
  __bf16* xab = (__bf16*)(ws + 8 * MB);     // activation staging (reused x3)  8 MB
  __bf16* qb  = (__bf16*)(ws + 16 * MB);    // [B,H,S,DH]                      8 MB
  __bf16* kb  = (__bf16*)(ws + 24 * MB);    // [B,H,S,DH]                      8 MB
  __bf16* vtb = (__bf16*)(ws + 32 * MB);    // [B,H,DH,SPc]                   ~8.25 MB
  const size_t vt_bytes = (size_t)Bc * Hc * DHc * SPc * sizeof(__bf16);
  __bf16* ctxb = (__bf16*)(ws + 32 * MB + vt_bytes);  // [B*S, D]              8 MB

  // zero V^T (covers the padded key tail read by attention)
  hipMemsetAsync(vtb, 0, vt_bytes, stream);

  const int nW = Dc * Dc;          // weight elements
  const int nX = Mc * Dc;          // activation elements
  const dim3 cvtW((nW + 255) / 256), cvtX((nX + 255) / 256);
  cvt_f32_bf16_kernel<<<cvtW, 256, 0, stream>>>(Wq, wqb, nW);
  cvt_f32_bf16_kernel<<<cvtW, 256, 0, stream>>>(Wk, wkb, nW);
  cvt_f32_bf16_kernel<<<cvtW, 256, 0, stream>>>(Wv, wvb, nW);
  cvt_f32_bf16_kernel<<<cvtW, 256, 0, stream>>>(Wo, wob, nW);

  const dim3 gGrid(Mc / 128, Dc / 64);  // (32,16), 4 waves/block, 32x64 per wave

  // Q projection
  cvt_f32_bf16_kernel<<<cvtX, 256, 0, stream>>>(Xq, xab, nX);
  gemm_kernel<0><<<gGrid, 128, 0, stream>>>(xab, wqb, bq, qb, nullptr);
  // K projection (reuse staging buffer; stream-ordered)
  cvt_f32_bf16_kernel<<<cvtX, 256, 0, stream>>>(Xk, xab, nX);
  gemm_kernel<0><<<gGrid, 128, 0, stream>>>(xab, wkb, bk, kb, nullptr);
  // V projection -> transposed V^T layout
  cvt_f32_bf16_kernel<<<cvtX, 256, 0, stream>>>(Xv, xab, nX);
  gemm_kernel<2><<<gGrid, 128, 0, stream>>>(xab, wvb, bv, vtb, nullptr);

  attn_kernel<<<dim3(Bc * Hc, Sc / 16), 32, 0, stream>>>(qb, kb, vtb, ctxb);

  // Output projection -> fp32 d_out
  gemm_kernel<3><<<gGrid, 128, 0, stream>>>(ctxb, wob, bo, nullptr, (float*)d_out);
}
